// MultiQuerySelfAttention_11450382811215
// MI455X (gfx1250) — matrix operand order repaired and checked by reading, not compiled, not run on hardware
//
#include <hip/hip_runtime.h>
#include <hip/hip_bf16.h>

typedef __attribute__((ext_vector_type(16))) __bf16 v16bf;
typedef __attribute__((ext_vector_type(8)))  __bf16 v8bf;
typedef __attribute__((ext_vector_type(8)))  float  v8f;

#define SHUF16 0,1,2,3,4,5,6,7,8,9,10,11,12,13,14,15

__device__ __forceinline__ unsigned short f32_to_bf16_bits(float f) {
    unsigned u = __builtin_bit_cast(unsigned, f);
    unsigned r = u + 0x7FFFu + ((u >> 16) & 1u);   // round-to-nearest-even
    return (unsigned short)(r >> 16);
}
__device__ __forceinline__ unsigned pack2(float a, float b) {
    return (unsigned)f32_to_bf16_bits(a) | ((unsigned)f32_to_bf16_bits(b) << 16);
}

// ---------------------------------------------------------------------------
// CDNA5 async global->LDS copy (ASYNCcnt-tracked) + wait helpers.
// Generic pointers to __shared__ carry the LDS byte offset in their low 32
// bits (flat LDS aperture), which is exactly the VDST
// operand of GLOBAL_LOAD_ASYNC_TO_LDS_*.
// ---------------------------------------------------------------------------
__device__ __forceinline__ void async_b128(const unsigned short* g, unsigned short* l) {
    unsigned loff = (unsigned)(unsigned long long)(uintptr_t)l;
    unsigned long long ga = (unsigned long long)(uintptr_t)g;
    asm volatile("global_load_async_to_lds_b128 %0, %1, off"
                 :: "v"(loff), "v"(ga) : "memory");
}
__device__ __forceinline__ void wait_async0() {
#if __has_builtin(__builtin_amdgcn_s_wait_asynccnt)
    __builtin_amdgcn_s_wait_asynccnt(0);
#else
    asm volatile("s_wait_asynccnt 0x0" ::: "memory");
#endif
}

// ---------------------------------------------------------------------------
// fp32 -> bf16 conversion
// ---------------------------------------------------------------------------
__global__ void f32_to_bf16_kernel(const float* __restrict__ in,
                                   unsigned short* __restrict__ out, int n) {
    int i = blockIdx.x * blockDim.x + threadIdx.x;
    if (i < n) out[i] = f32_to_bf16_bits(in[i]);
}

// ---------------------------------------------------------------------------
// V [B*L, dh] -> Vt [B, dh, L]
// ---------------------------------------------------------------------------
__global__ void transpose_v_kernel(const unsigned short* __restrict__ V,
                                   unsigned short* __restrict__ Vt,
                                   int L, int n) {
    int idx = blockIdx.x * blockDim.x + threadIdx.x;   // b*128*L + d*L + l
    if (idx >= n) return;
    int l = idx % L;
    int rest = idx / L;
    int d = rest & 127;
    int b = rest >> 7;
    Vt[idx] = V[(size_t)(b * L + l) * 128 + d];
}

// ---------------------------------------------------------------------------
// GEMM: C[M,N] = A[M,K] @ W[N,K]^T  (bf16 in, fp32 accumulate)
// 256 thr = 8 waves; block tile 128(M) x 128(N); wave tile 64 x 32.
// K-chunks of 32 double-buffered in LDS, filled with async-to-LDS loads.
// Requires M % 128 == 0, N % 128 == 0, K % 32 == 0.
// ---------------------------------------------------------------------------
template <bool OUT_F32, bool HAS_BIAS>
__global__ __launch_bounds__(256)
void gemm_bf16_nt(const unsigned short* __restrict__ A,
                  const unsigned short* __restrict__ W,
                  void* __restrict__ Cp,
                  const float* __restrict__ bias,
                  int M, int N, int K) {
    // row stride 48 halves = 96B = 3*32B -> every row 32B-aligned
    __shared__ __attribute__((aligned(32))) unsigned short As[2][128 * 48];
    __shared__ __attribute__((aligned(32))) unsigned short Bs[2][128 * 48];

    const int lane = threadIdx.x & 31;
    const int wid  = threadIdx.x >> 5;       // 0..7
    const int l15  = lane & 15;
    const int hi   = lane >> 4;              // 0 or 1

    const int mb  = blockIdx.y * 128;
    const int nbb = blockIdx.x * 128;
    const int mo  = (wid >> 2) * 64;         // wave M offset: 0 / 64
    const int no  = (wid & 3) * 32;          // wave N offset: 0/32/64/96

    // staging role: 256 threads, one 64B row-chunk each (A: rows 0..127, B: rows 0..127)
    const int r = threadIdx.x & 127;
    const int halfSel = threadIdx.x >> 7;    // 0 -> A, 1 -> B
    const unsigned short* srow = halfSel ? (W + (size_t)(nbb + r) * K)
                                         : (A + (size_t)(mb + r) * K);

    auto stage = [&](int buf, int k0) {
        unsigned short* drow = (halfSel ? &Bs[buf][0] : &As[buf][0]) + r * 48;
#pragma unroll
        for (int j = 0; j < 4; ++j)
            async_b128(srow + k0 + j * 8, drow + j * 8);
    };

    v8f c[4][2];
    v8f z = {};
#pragma unroll
    for (int mt = 0; mt < 4; ++mt) { c[mt][0] = z; c[mt][1] = z; }

    stage(0, 0);

    for (int k0 = 0; k0 < K; k0 += 32) {
        const int cur = (k0 >> 5) & 1;
        wait_async0();
        __syncthreads();
        if (k0 + 32 < K) stage(cur ^ 1, k0 + 32);

        const unsigned short* as = &As[cur][0];
        const unsigned short* bs = &Bs[cur][0];
        // B operands: lane n = l15, 16 contiguous K halves at hi*16
        v16bf bt0 = *reinterpret_cast<const v16bf*>(bs + (no + l15) * 48 + hi * 16);
        v16bf bt1 = *reinterpret_cast<const v16bf*>(bs + (no + 16 + l15) * 48 + hi * 16);
#pragma unroll
        for (int mt = 0; mt < 4; ++mt) {
            const unsigned short* ap = as + (mo + mt * 16 + l15) * 48 + hi * 16;
            v8bf alo = *reinterpret_cast<const v8bf*>(ap);
            v8bf ahi = *reinterpret_cast<const v8bf*>(ap + 8);
            v16bf av = __builtin_shufflevector(alo, ahi, SHUF16);
            c[mt][0] = __builtin_amdgcn_wmma_f32_16x16x32_bf16(false, av, false, bt0, (short)0, c[mt][0], false, false);
            c[mt][1] = __builtin_amdgcn_wmma_f32_16x16x32_bf16(false, av, false, bt1, (short)0, c[mt][1], false, false);
        }
    }

    // C layout per tile: lane col n = l15; rows m = v + hi*8
#pragma unroll
    for (int mt = 0; mt < 4; ++mt) {
#pragma unroll
        for (int nt = 0; nt < 2; ++nt) {
            const int ncol = nbb + no + nt * 16 + l15;
            const int mrow = mb + mo + mt * 16 + hi * 8;
            float bv = 0.f;
            if constexpr (HAS_BIAS) bv = bias[ncol];
            if constexpr (OUT_F32) {
                float* C = (float*)Cp;
#pragma unroll
                for (int v = 0; v < 8; ++v)
                    C[(size_t)(mrow + v) * N + ncol] = c[mt][nt][v] + bv;
            } else {
                unsigned short* C = (unsigned short*)Cp;
#pragma unroll
                for (int v = 0; v < 8; ++v)
                    C[(size_t)(mrow + v) * N + ncol] = f32_to_bf16_bits(c[mt][nt][v]);
            }
        }
    }
}

// ---------------------------------------------------------------------------
// Flash attention (MQA): block = 4 waves, same (b,h), 64 queries (16/wave).
// Keys streamed in chunks of 32, double-buffered in LDS via async loads.
// S^T = K.Q^T (softmax axis in VGPR dim), O^T = Vt.P^T with P^T relaid
// through per-wave LDS.
// ---------------------------------------------------------------------------
__global__ __launch_bounds__(128)
void attn_kernel(const unsigned short* __restrict__ Qbf,  // [B*L, 2048]
                 const unsigned short* __restrict__ Kbf,  // [B*L, 128]
                 const unsigned short* __restrict__ Vt,   // [B, 128, L]
                 unsigned short* __restrict__ Obf,        // [B*L, 2048]
                 int L) {
    __shared__ __attribute__((aligned(32))) unsigned short Ks[2][32 * 136];   // 32k x 128d (pad 8)
    __shared__ __attribute__((aligned(32))) unsigned short Vts[2][128 * 40];  // 128d x 32k (pad 8)
    __shared__ __attribute__((aligned(32))) unsigned short Pt[4][16 * 32];    // per-wave P^T relay

    const int lane = threadIdx.x & 31;
    const int wid  = threadIdx.x >> 5;
    const int l15  = lane & 15;
    const int hi   = lane >> 4;

    const int bh = blockIdx.y;         // B*H
    const int b  = bh >> 4;
    const int h  = bh & 15;
    const int q0 = blockIdx.x * 64 + wid * 16;
    const float scale = 0.08838834764831845f;  // 1/sqrt(128)

    // Q resident in registers as WMMA B operands: lane n=q=l15, K(d) halves hi*16..+15
    v16bf qreg[4];
    {
        const unsigned short* qrow =
            Qbf + (size_t)(b * L + q0 + l15) * 2048 + h * 128 + hi * 16;
#pragma unroll
        for (int dc = 0; dc < 4; ++dc)
            qreg[dc] = *reinterpret_cast<const v16bf*>(qrow + dc * 32);
    }

    auto stage = [&](int buf, int k0) {
        {   // K chunk: 32 rows x 128 cols, 64B per thread
            int r = threadIdx.x >> 2, seg = threadIdx.x & 3;
            const unsigned short* s = Kbf + (size_t)(b * L + k0 + r) * 128 + seg * 32;
            unsigned short* d = &Ks[buf][r * 136 + seg * 32];
#pragma unroll
            for (int j = 0; j < 4; ++j) async_b128(s + j * 8, d + j * 8);
        }
        {   // Vt chunk: 128 rows x 32 cols, one 64B row per thread
            int dr = threadIdx.x;
            const unsigned short* s = Vt + ((size_t)b * 128 + dr) * L + k0;
            unsigned short* d = &Vts[buf][dr * 40];
#pragma unroll
            for (int j = 0; j < 4; ++j) async_b128(s + j * 8, d + j * 8);
        }
    };

    v8f o[8];
    v8f z = {};
#pragma unroll
    for (int i = 0; i < 8; ++i) o[i] = z;
    float m_run = -3.0e38f;
    float l_run = 0.0f;

    stage(0, 0);

    for (int k0 = 0; k0 < L; k0 += 32) {
        const int cur = (k0 >> 5) & 1;
        wait_async0();
        __syncthreads();
        if (k0 + 32 < L) stage(cur ^ 1, k0 + 32);

        const unsigned short* ks = &Ks[cur][0];
        const unsigned short* vs = &Vts[cur][0];

        // --- S^T tiles: [16k x 16q] for key sub-tiles kt=0,16 ---
        v8f s0 = z, s1 = z;
#pragma unroll
        for (int dc = 0; dc < 4; ++dc) {
            const unsigned short* a0p = ks + l15 * 136 + dc * 32 + hi * 16;
            v8bf a0lo = *reinterpret_cast<const v8bf*>(a0p);
            v8bf a0hi = *reinterpret_cast<const v8bf*>(a0p + 8);
            v16bf a0 = __builtin_shufflevector(a0lo, a0hi, SHUF16);
            s0 = __builtin_amdgcn_wmma_f32_16x16x32_bf16(false, a0, false, qreg[dc], (short)0, s0, false, false);
            const unsigned short* a1p = ks + (16 + l15) * 136 + dc * 32 + hi * 16;
            v8bf a1lo = *reinterpret_cast<const v8bf*>(a1p);
            v8bf a1hi = *reinterpret_cast<const v8bf*>(a1p + 8);
            v16bf a1 = __builtin_shufflevector(a1lo, a1hi, SHUF16);
            s1 = __builtin_amdgcn_wmma_f32_16x16x32_bf16(false, a1, false, qreg[dc], (short)0, s1, false, false);
        }

        // --- online softmax: key index in VGPR dim + half-wave dim ---
        float t0[8], t1[8];
        float mpart = -3.0e38f;
#pragma unroll
        for (int v = 0; v < 8; ++v) {
            t0[v] = s0[v] * scale;
            t1[v] = s1[v] * scale;
            mpart = fmaxf(mpart, fmaxf(t0[v], t1[v]));
        }
        float mblk = fmaxf(mpart, __shfl_xor(mpart, 16, 32));
        float mnew = fmaxf(m_run, mblk);
        float alpha = __expf(m_run - mnew);
        float lpart = 0.f;
#pragma unroll
        for (int v = 0; v < 8; ++v) {
            t0[v] = __expf(t0[v] - mnew);
            t1[v] = __expf(t1[v] - mnew);
            lpart += t0[v] + t1[v];
        }
        l_run = alpha * l_run + lpart + __shfl_xor(lpart, 16, 32);
        m_run = mnew;
#pragma unroll
        for (int i = 0; i < 8; ++i) o[i] *= alpha;

        // --- relay P^T through per-wave LDS: Pt[q][k] ---
        {
            unsigned short* pw = Pt[wid];
            uint4 p0, p1;
            p0.x = pack2(t0[0], t0[1]); p0.y = pack2(t0[2], t0[3]);
            p0.z = pack2(t0[4], t0[5]); p0.w = pack2(t0[6], t0[7]);
            p1.x = pack2(t1[0], t1[1]); p1.y = pack2(t1[2], t1[3]);
            p1.z = pack2(t1[4], t1[5]); p1.w = pack2(t1[6], t1[7]);
            *reinterpret_cast<uint4*>(pw + l15 * 32 + hi * 8)      = p0;  // k = hi*8+v
            *reinterpret_cast<uint4*>(pw + l15 * 32 + 16 + hi * 8) = p1;  // k = 16+hi*8+v
        }
        v16bf pb = *reinterpret_cast<const v16bf*>(Pt[wid] + l15 * 32 + hi * 16);

        // --- O^T += Vt_tile . P^T : 8 d-tiles of 16 ---
#pragma unroll
        for (int dt = 0; dt < 8; ++dt) {
            const unsigned short* vp = vs + (dt * 16 + l15) * 40 + hi * 16;
            v8bf vlo = *reinterpret_cast<const v8bf*>(vp);
            v8bf vhi = *reinterpret_cast<const v8bf*>(vp + 8);
            v16bf va = __builtin_shufflevector(vlo, vhi, SHUF16);
            o[dt] = __builtin_amdgcn_wmma_f32_16x16x32_bf16(false, va, false, pb, (short)0, o[dt], false, false);
        }
    }

    // --- normalize and store: lane col q=l15, rows d = dt*16 + hi*8 + v ---
    float rl = 1.0f / l_run;
    unsigned short* orow =
        Obf + (size_t)(b * L + q0 + l15) * 2048 + h * 128 + hi * 8;
#pragma unroll
    for (int dt = 0; dt < 8; ++dt) {
        uint4 pk;
        pk.x = pack2(o[dt][0] * rl, o[dt][1] * rl);
        pk.y = pack2(o[dt][2] * rl, o[dt][3] * rl);
        pk.z = pack2(o[dt][4] * rl, o[dt][5] * rl);
        pk.w = pack2(o[dt][6] * rl, o[dt][7] * rl);
        *reinterpret_cast<uint4*>(orow + dt * 16) = pk;
    }
}

// ---------------------------------------------------------------------------
extern "C" void kernel_launch(void* const* d_in, const int* in_sizes, int n_in,
                              void* d_out, int out_size, void* d_ws, size_t ws_size,
                              hipStream_t stream) {
    const float* x  = (const float*)d_in[0];
    const float* Wq = (const float*)d_in[1];
    const float* Wk = (const float*)d_in[2];
    const float* Wv = (const float*)d_in[3];
    const float* Wo = (const float*)d_in[4];
    const float* bo = (const float*)d_in[5];
    float* out = (float*)d_out;

    const int B = 2, L = 2048, D = 2048, H = 16, dh = 128;
    const int M = B * L;  // 4096

    char* ws = (char*)d_ws;
    unsigned short* xbf = (unsigned short*)ws; ws += (size_t)M * D * 2;
    unsigned short* Wqb = (unsigned short*)ws; ws += (size_t)D * D * 2;
    unsigned short* Wkb = (unsigned short*)ws; ws += (size_t)dh * D * 2;
    unsigned short* Wvb = (unsigned short*)ws; ws += (size_t)dh * D * 2;
    unsigned short* Wob = (unsigned short*)ws; ws += (size_t)D * D * 2;
    unsigned short* Qb  = (unsigned short*)ws; ws += (size_t)M * D * 2;
    unsigned short* Kb  = (unsigned short*)ws; ws += (size_t)M * dh * 2;
    unsigned short* Vb  = (unsigned short*)ws; ws += (size_t)M * dh * 2;
    unsigned short* Vtb = (unsigned short*)ws; ws += (size_t)M * dh * 2;
    unsigned short* Ab  = (unsigned short*)ws; ws += (size_t)M * D * 2;

    auto cvt = [&](const float* s, unsigned short* d, int n) {
        f32_to_bf16_kernel<<<(n + 255) / 256, 256, 0, stream>>>(s, d, n);
    };
    cvt(x,  xbf, M * D);
    cvt(Wq, Wqb, D * D);
    cvt(Wk, Wkb, dh * D);
    cvt(Wv, Wvb, dh * D);
    cvt(Wo, Wob, D * D);

    dim3 blk(256);
    // Q = x @ Wq^T  [4096 x 2048]
    gemm_bf16_nt<false, false><<<dim3(D / 128, M / 128), blk, 0, stream>>>(xbf, Wqb, Qb, nullptr, M, D, D);
    // K = x @ Wk^T  [4096 x 128]
    gemm_bf16_nt<false, false><<<dim3(dh / 128, M / 128), blk, 0, stream>>>(xbf, Wkb, Kb, nullptr, M, dh, D);
    // V = x @ Wv^T  [4096 x 128]
    gemm_bf16_nt<false, false><<<dim3(dh / 128, M / 128), blk, 0, stream>>>(xbf, Wvb, Vb, nullptr, M, dh, D);
    // Vt[b,d,l]
    {
        int n = B * dh * L;
        transpose_v_kernel<<<(n + 255) / 256, 256, 0, stream>>>(Vb, Vtb, L, n);
    }
    // attention
    attn_kernel<<<dim3(L / 64, B * H), dim3(128), 0, stream>>>(Qb, Kb, Vtb, Ab, L);
    // out = A @ Wo^T + bo   (fp32 out)
    gemm_bf16_nt<true, true><<<dim3(D / 128, M / 128), blk, 0, stream>>>(Ab, Wob, out, bo, M, D, D);
}
